// Model_39676907887797
// MI455X (gfx1250) — matrix operand order repaired and checked by reading, not compiled, not run on hardware
//
#include <hip/hip_runtime.h>
#include <hip/hip_bf16.h>

typedef __attribute__((ext_vector_type(16))) __bf16 v16bf;
typedef __attribute__((ext_vector_type(8)))  float  v8f;

#define WMMA_BF16(A, Bm, C) \
    __builtin_amdgcn_wmma_f32_16x16x32_bf16(false, (A), false, (Bm), (short)0, (C), false, false)

namespace {
constexpr int kB  = 16;
constexpr int kSQ = 2048;
constexpr int kSK = 2048;
constexpr int kD  = 128;
constexpr int kDV = 128;

constexpr int BR = 128;           // query rows per block (8 waves * 16)
constexpr int BC = 64;            // key rows per iteration
constexpr int NT = kSK / BC;      // 32 iterations
}

__device__ __forceinline__ v8f vzero8() {
    v8f z;
#pragma unroll
    for (int i = 0; i < 8; ++i) z[i] = 0.0f;
    return z;
}

// Exchange a 32-bit value with lane^16 (SWAPX16 pattern, group-of-32 mode:
// and=0x1f, or=0, xor=0x10 -> offset 0x401f). DS unit, no memory traffic.
__device__ __forceinline__ int swap16_i32(int x) {
    return __builtin_amdgcn_ds_swizzle(x, 0x401f);
}
__device__ __forceinline__ float swap16_f32(float x) {
    return __builtin_bit_cast(float, swap16_i32(__builtin_bit_cast(int, x)));
}

// Pack two f32 into one dword of two bf16 (lo, hi).
__device__ __forceinline__ int pack_bf16(float lo, float hi) {
    unsigned short ls = __builtin_bit_cast(unsigned short, (__bf16)lo);
    unsigned short hs = __builtin_bit_cast(unsigned short, (__bf16)hi);
    return (int)(((unsigned)hs << 16) | (unsigned)ls);
}

__global__ __launch_bounds__(256, 1)
void fattn_fwd(const float* __restrict__ Q,
               const float* __restrict__ K,
               const float* __restrict__ V,
               const float* __restrict__ scale_p,
               float* __restrict__ O) {
    __shared__ __bf16 kbuf[BC * kD];      // 64*128*2B = 16 KB (row-major k x d)
    __shared__ __bf16 vtbuf[kDV * BC];    // 128*64*2B = 16 KB (transposed dv x k)

    const int tid   = threadIdx.x;
    const int lane  = tid & 31;
    const int lhalf = lane & 15;
    const bool hi   = (lane >= 16);
    const int wave  = tid >> 5;

    const int b     = blockIdx.x / (kSQ / BR);
    const int qtile = blockIdx.x % (kSQ / BR);
    const int qrow0 = qtile * BR + wave * 16;

    // scale folded with log2(e): softmax runs in exp2 domain.
    const float sl2 = scale_p[0] * 1.44269504088896340736f;

    const float* Qb = Q + (size_t)b * kSQ * kD;
    const float* Kb = K + (size_t)b * kSK * kD;
    const float* Vb = V + (size_t)b * kSK * kDV;
    float*       Ob = O + (size_t)b * kSQ * kDV;

    // ---- Q as B-operand fragments (Sᵀ = K·Qᵀ). B layout: lane owns column
    // q = lhalf; lanes 0-15 carry d = base+0..15, lanes 16-31 d = base+16..31,
    // elements sequential -> 16 consecutive floats of the Q row.
    v16bf qb[4];
    {
        const float* qr = Qb + (size_t)(qrow0 + lhalf) * kD;
#pragma unroll
        for (int c = 0; c < 4; ++c) {
            const int d0 = c * 32 + (hi ? 16 : 0);
#pragma unroll
            for (int e = 0; e < 16; ++e) qb[c][e] = (__bf16)qr[d0 + e];
        }
    }

    // ---- Per-lane softmax state (lane = query row) + Oᵀ accumulators:
    // 8 C tiles, each 16(dv) x 16(q); lane = q, VGPR = dv row.
    float mrow = -3.0e38f, lrow = 0.0f;
    v8f oaccT[8];
#pragma unroll
    for (int t = 0; t < 8; ++t) oaccT[t] = vzero8();

    // ---- Stage K/V tile 0 into LDS (fp32 -> bf16, V transposed). ----
#pragma unroll
    for (int u = 0; u < 8; ++u) {
        const int idx = (tid + u * 256) * 4;
        float4 fk = *(const float4*)(Kb + idx);
        __bf16* kd = &kbuf[idx];
        kd[0] = (__bf16)fk.x; kd[1] = (__bf16)fk.y;
        kd[2] = (__bf16)fk.z; kd[3] = (__bf16)fk.w;

        float4 fv = *(const float4*)(Vb + idx);
        const int k  = idx / kDV;
        const int dv = idx % kDV;
        vtbuf[(dv + 0) * BC + k] = (__bf16)fv.x;
        vtbuf[(dv + 1) * BC + k] = (__bf16)fv.y;
        vtbuf[(dv + 2) * BC + k] = (__bf16)fv.z;
        vtbuf[(dv + 3) * BC + k] = (__bf16)fv.w;
    }
    __syncthreads();

    for (int it = 0; it < NT; ++it) {
        const bool havenxt = (it + 1 < NT);

        // -- Next tile's global loads issue now, complete during the WMMAs. --
        float4 kreg[8], vreg[8];
        if (havenxt) {
            const float* kp = Kb + (size_t)(it + 1) * BC * kD;
            const float* vp = Vb + (size_t)(it + 1) * BC * kDV;
#pragma unroll
            for (int u = 0; u < 8; ++u) {
                kreg[u] = *(const float4*)(kp + (tid + u * 256) * 4);
                vreg[u] = *(const float4*)(vp + (tid + u * 256) * 4);
            }
        }
        if (it + 2 < NT) {
            __builtin_prefetch(Kb + (size_t)(it + 2) * BC * kD  + tid * 32, 0, 3);
            __builtin_prefetch(Vb + (size_t)(it + 2) * BC * kDV + tid * 32, 0, 3);
        }

        // ---- Sᵀ = K · Qᵀ : four 16(k) x 16(q) C tiles per wave. ----
        // K as A-operand: lane owns k-row j*16+lhalf; lo lanes d=0..15,
        // hi lanes d=16..31 within each 32-wide d chunk.
        v8f sT[4];
#pragma unroll
        for (int j = 0; j < 4; ++j) {
            v16bf kf[4];
#pragma unroll
            for (int c = 0; c < 4; ++c) {
                const __bf16* kr = kbuf + (j * 16 + lhalf) * kD +
                                   c * 32 + (hi ? 16 : 0);
#pragma unroll
                for (int e = 0; e < 8; ++e) {
                    kf[c][e]     = kr[e];
                    kf[c][8 + e] = kr[8 + e];
                }
            }
            v8f s = vzero8();
#pragma unroll
            for (int c = 0; c < 4; ++c) s = WMMA_BF16(kf[c], qb[c], s);
            sT[j] = s;
        }

        // ---- Online softmax: per-lane over 32 k-values + one xor-16 combine. --
#pragma unroll
        for (int j = 0; j < 4; ++j) sT[j] *= sl2;   // into exp2 domain

        float m = sT[0][0];
#pragma unroll
        for (int j = 0; j < 4; ++j)
#pragma unroll
            for (int r = 0; r < 8; ++r) m = fmaxf(m, sT[j][r]);
        m = fmaxf(m, swap16_f32(m));                // lanes L and L^16 share q

        const float mnew = fmaxf(mrow, m);
        const float fac  = __builtin_amdgcn_exp2f(mrow - mnew);
        mrow = mnew;

        float rs = 0.0f;
#pragma unroll
        for (int j = 0; j < 4; ++j)
#pragma unroll
            for (int r = 0; r < 8; ++r) {
                const float p = __builtin_amdgcn_exp2f(sT[j][r] - mnew);
                sT[j][r] = p;
                rs += p;
            }
        rs += swap16_f32(rs);
        lrow = lrow * fac + rs;

#pragma unroll
        for (int t = 0; t < 8; ++t) oaccT[t] *= fac;   // per-lane uniform fac

        // ---- Build Pᵀ B-operand fragments in registers. ----
        // Tile j holds k rows j*16+0..7 (lo lanes) / +8..15 (hi lanes).
        // B frag kc needs lo lanes: k=kc*32+0..15, hi lanes: k=kc*32+16..31.
        // Missing half comes from lane^16 via 4 packed ds_swizzle swaps.
        int pk[4][4];
#pragma unroll
        for (int j = 0; j < 4; ++j)
#pragma unroll
            for (int v = 0; v < 4; ++v)
                pk[j][v] = pack_bf16(sT[j][2 * v], sT[j][2 * v + 1]);

        v16bf pfB[2];
#pragma unroll
        for (int kc = 0; kc < 2; ++kc) {
            const int ja = 2 * kc, jb = 2 * kc + 1;
            union { int i[8]; v16bf v; } u;
#pragma unroll
            for (int v = 0; v < 4; ++v) {
                const int send = hi ? pk[ja][v] : pk[jb][v];
                const int recv = swap16_i32(send);
                u.i[v]     = hi ? recv      : pk[ja][v];  // k 0..7   / 16..23
                u.i[4 + v] = hi ? pk[jb][v] : recv;       // k 8..15  / 24..31
            }
            pfB[kc] = u.v;
        }

        // ---- Oᵀ += Vᵀ · Pᵀ : A = Vᵀ tiles from LDS, B = Pᵀ from registers. --
#pragma unroll
        for (int t = 0; t < 8; ++t) {
#pragma unroll
            for (int kc = 0; kc < 2; ++kc) {
                const __bf16* vr = vtbuf + (t * 16 + lhalf) * BC +
                                   kc * 32 + (hi ? 16 : 0);
                v16bf vf;
#pragma unroll
                for (int e = 0; e < 8; ++e) {
                    vf[e]     = vr[e];
                    vf[8 + e] = vr[8 + e];
                }
                oaccT[t] = WMMA_BF16(vf, pfB[kc], oaccT[t]);
            }
        }

        // ---- Commit the prefetched tile into LDS (single buffer). ----
        __syncthreads();
        if (havenxt) {
#pragma unroll
            for (int u = 0; u < 8; ++u) {
                const int idx = (tid + u * 256) * 4;
                __bf16* kd = &kbuf[idx];
                kd[0] = (__bf16)kreg[u].x; kd[1] = (__bf16)kreg[u].y;
                kd[2] = (__bf16)kreg[u].z; kd[3] = (__bf16)kreg[u].w;
                const int k  = idx / kDV;
                const int dv = idx % kDV;
                vtbuf[(dv + 0) * BC + k] = (__bf16)vreg[u].x;
                vtbuf[(dv + 1) * BC + k] = (__bf16)vreg[u].y;
                vtbuf[(dv + 2) * BC + k] = (__bf16)vreg[u].z;
                vtbuf[(dv + 3) * BC + k] = (__bf16)vreg[u].w;
            }
        }
        __syncthreads();
    }

    // ---- Epilogue: per-lane normalize, store O[q][dv] (fp32). ----
    {
        const float inv = 1.0f / lrow;
        float* orow = Ob + (size_t)(qrow0 + lhalf) * kDV;
#pragma unroll
        for (int t = 0; t < 8; ++t) {
#pragma unroll
            for (int r = 0; r < 8; ++r) {
                const int dv = t * 16 + (hi ? (r + 8) : r);
                orow[dv] = oaccT[t][r] * inv;
            }
        }
    }
}

extern "C" void kernel_launch(void* const* d_in, const int* in_sizes, int n_in,
                              void* d_out, int out_size, void* d_ws, size_t ws_size,
                              hipStream_t stream) {
    (void)in_sizes; (void)n_in; (void)out_size; (void)d_ws; (void)ws_size;
    const float* Q     = (const float*)d_in[0];
    const float* K     = (const float*)d_in[1];
    const float* V     = (const float*)d_in[2];
    const float* scale = (const float*)d_in[4];   // d_in[3] (x4) unused by forward
    float* O = (float*)d_out;

    dim3 grid(kB * (kSQ / BR));   // 256 blocks
    dim3 block(256);              // 8 waves (wave32)
    fattn_fwd<<<grid, block, 0, stream>>>(Q, K, V, scale, O);
}
